// attention_head_55379308315467
// MI455X (gfx1250) — matrix operand order repaired and checked by reading, not compiled, not run on hardware
//
#include <hip/hip_runtime.h>

// ---------------------------------------------------------------------------
// MI455X (gfx1250) single-head attention, bf16 WMMA flash-attention pipeline.
// Roofline: ~26 GFLOP vs ~330MB HBM traffic -> memory bound at 23.3 TB/s.
// So: one pass over q/k/v, no materialized SxS scores, bf16 matrix cores.
// ---------------------------------------------------------------------------

typedef __attribute__((ext_vector_type(16))) __bf16 v16bf;
typedef __attribute__((ext_vector_type(8)))  __bf16 v8bf;
typedef __attribute__((ext_vector_type(8)))  float  v8f;
typedef __attribute__((ext_vector_type(4)))  float  v4f;

constexpr int kDModel = 1024;
constexpr int kDK     = 64;
constexpr int kSeq    = 4096;
constexpr int kBatch  = 4;
constexpr int kRows   = kBatch * kSeq;   // 16384 flattened (b,s) rows

__device__ __forceinline__ v8f wmma_bf16(v16bf a, v16bf b, v8f c) {
  // D = A(16x32) * B(32x16) + C(16x16), f32 accumulate
  return __builtin_amdgcn_wmma_f32_16x16x32_bf16(false, a, false, b, (short)0, c,
                                                 false, false);
}

__device__ __forceinline__ v16bf pack2(v8bf lo, v8bf hi8) {
  v16bf r;
#pragma unroll
  for (int i = 0; i < 8; ++i) { r[i] = lo[i]; r[8 + i] = hi8[i]; }
  return r;
}

// A fragment (16x32 bf16) from an f32 row pointer (already at k0).
// ISA layout: lane row = lane&15; e<8 -> K=kb+e, e>=8 -> K=16+kb+(e-8), kb=8*(lane>=16)
__device__ __forceinline__ v16bf a_frag_f32(const float* p, int kb) {
  v4f f0 = *(const v4f*)(p + kb);
  v4f f1 = *(const v4f*)(p + kb + 4);
  v4f f2 = *(const v4f*)(p + 16 + kb);
  v4f f3 = *(const v4f*)(p + 16 + kb + 4);
  v16bf r;
#pragma unroll
  for (int i = 0; i < 4; ++i) {
    r[i]      = (__bf16)f0[i];
    r[4 + i]  = (__bf16)f1[i];
    r[8 + i]  = (__bf16)f2[i];
    r[12 + i] = (__bf16)f3[i];
  }
  return r;
}

// B fragment: f32 at p..p+7 and p+16..p+23 (pointer already offset by 8*(lane>=16)).
__device__ __forceinline__ v16bf b_frag_f32(const float* p) {
  v4f f0 = *(const v4f*)(p);
  v4f f1 = *(const v4f*)(p + 4);
  v4f f2 = *(const v4f*)(p + 16);
  v4f f3 = *(const v4f*)(p + 20);
  v16bf r;
#pragma unroll
  for (int i = 0; i < 4; ++i) {
    r[i]      = (__bf16)f0[i];
    r[4 + i]  = (__bf16)f1[i];
    r[8 + i]  = (__bf16)f2[i];
    r[12 + i] = (__bf16)f3[i];
  }
  return r;
}

__device__ __forceinline__ v16bf a_frag_bf16(const __bf16* p, int kb) {
  return pack2(*(const v8bf*)(p + kb), *(const v8bf*)(p + 16 + kb));
}
__device__ __forceinline__ v16bf b_frag_bf16(const __bf16* p) {
  return pack2(*(const v8bf*)(p), *(const v8bf*)(p + 16));
}

// ---------------------------------------------------------------------------
// Stage 1: qh = (q@Wq^T + bq) * 0.125*log2e  (scale+log2e folded for exp2 softmax)
//          kh = k@Wk^T + bk          (row-major bf16 [16384,64])
//          vhT = (v@Wv^T + bv)^T     (bf16 [64,16384]; contiguous-K B frags later)
// One 16-row tile per wave; 3072 wave-jobs.
// ---------------------------------------------------------------------------
__global__ __launch_bounds__(256) void proj_wmma(
    const float* __restrict__ qx, const float* __restrict__ kx,
    const float* __restrict__ vx,
    const float* __restrict__ Wq, const float* __restrict__ bq,
    const float* __restrict__ Wk, const float* __restrict__ bk,
    const float* __restrict__ Wv, const float* __restrict__ bv,
    __bf16* __restrict__ qh, __bf16* __restrict__ kh, __bf16* __restrict__ vhT) {
  const int lane = threadIdx.x & 31;
  const int wv   = threadIdx.x >> 5;
  const int job  = blockIdx.x * 8 + wv;     // 0..3071
  const int which = job >> 10;              // 0=q, 1=k, 2=v
  const int m0   = (job & 1023) * 16;       // row-tile base
  const int row  = lane & 15;
  const int hi   = lane >> 4;
  const int kbA  = hi * 8;
  const int kbB  = hi * 8;

  const float* x; const float* W; const float* bias; float scale;
  if (which == 0)      { x = qx; W = Wq; bias = bq; scale = 0.125f * 1.44269504088896f; }
  else if (which == 1) { x = kx; W = Wk; bias = bk; scale = 1.0f; }
  else                 { x = vx; W = Wv; bias = bv; scale = 1.0f; }

  v8f acc[4] = {};
  const float* xr = x + (size_t)(m0 + row) * kDModel;

  for (int k0 = 0; k0 < kDModel; k0 += 32) {
    v16bf A = a_frag_f32(xr + k0, kbA);
#pragma unroll
    for (int j = 0; j < 4; ++j) {
      // B[k,n] = W[n, k]; lane holds column n = j*16+row, contiguous K
      v16bf B = b_frag_f32(W + (size_t)(j * 16 + row) * kDModel + k0 + kbB);
      acc[j] = wmma_bf16(A, B, acc[j]);
    }
  }

#pragma unroll
  for (int j = 0; j < 4; ++j) {
    const int col = j * 16 + row;             // D layout: N = lane&15
    const float bval = bias[col];
#pragma unroll
    for (int r = 0; r < 8; ++r) {             // D layout: M = r + 8*hi
      const int orow = m0 + r + hi * 8;
      const float y = (acc[j][r] + bval) * scale;
      if (which == 2)       vhT[(size_t)col * kRows + orow] = (__bf16)y;
      else if (which == 0)  qh[(size_t)orow * kDK + col]    = (__bf16)y;
      else                  kh[(size_t)orow * kDK + col]    = (__bf16)y;
    }
  }
}

// ---------------------------------------------------------------------------
// Stage 2: flash attention. One (batch, 16-query tile) per wave; 1024 jobs.
// 32 keys per iteration: 4 WMMA for scores (K=64) + 4 WMMA for P@V (K=32).
// Row-wise softmax reductions via 16-lane shfl_xor; P reshaped C->A layout
// through a per-wave LDS tile (stride 36 floats: 16B aligned, conflict-free).
// ---------------------------------------------------------------------------
__global__ __launch_bounds__(256) void flash_attn_wmma(
    const __bf16* __restrict__ qh, const __bf16* __restrict__ kh,
    const __bf16* __restrict__ vhT, const int* __restrict__ mask,
    __bf16* __restrict__ obf) {
  __shared__ float Pbuf[8][16 * 36];
  const int lane = threadIdx.x & 31;
  const int wv   = threadIdx.x >> 5;
  const int job  = blockIdx.x * 8 + wv;     // 0..1023
  const int b    = job >> 8;
  const int q0   = (job & 255) * 16;
  const int row  = lane & 15;
  const int hi   = lane >> 4;
  const int kbA  = hi * 8;
  const int kbB  = hi * 8;
  float* P = &Pbuf[wv][0];

  // Resident query A-fragments (qh already carries 0.125*log2e)
  const __bf16* qr = qh + (size_t)(b * kSeq + q0 + row) * kDK;
  const v16bf aq0 = a_frag_bf16(qr, kbA);        // K = 0..31
  const v16bf aq1 = a_frag_bf16(qr + 32, kbA);   // K = 32..63

  float m_i[8], l_i[8];
  v8f O[4] = {};
#pragma unroll
  for (int r = 0; r < 8; ++r) { m_i[r] = -1e30f; l_i[r] = 0.0f; }

  const __bf16* khb = kh + (size_t)b * kSeq * kDK;

  for (int kt = 0; kt < kSeq; kt += 32) {
    if (kt + 32 < kSeq) {   // stream-ahead hints for the K/V tiles
      __builtin_prefetch(khb + (size_t)(kt + 32 + row) * kDK, 0, 0);
      __builtin_prefetch(vhT + (size_t)row * kRows + b * kSeq + kt + 32, 0, 0);
    }

    // scores: s = qh(16x64) @ kh^T(64x32), two 16-col WMMA chains
    v8f s0 = {}, s1 = {};
    {
      const __bf16* kr0 = khb + (size_t)(kt + row) * kDK + kbB;
      s0 = wmma_bf16(aq0, b_frag_bf16(kr0), s0);
      s0 = wmma_bf16(aq1, b_frag_bf16(kr0 + 32), s0);
      const __bf16* kr1 = khb + (size_t)(kt + 16 + row) * kDK + kbB;
      s1 = wmma_bf16(aq0, b_frag_bf16(kr1), s1);
      s1 = wmma_bf16(aq1, b_frag_bf16(kr1 + 32), s1);
    }

    // mask + running row max (rows live across each 16-lane half at element r)
    float mnew[8];
#pragma unroll
    for (int r = 0; r < 8; ++r) {
      const int qrow = q0 + r + hi * 8;
      const int* mp = mask + (size_t)qrow * kSeq + kt;
      float a = (mp[row] != 0)      ? s0[r] : -1e30f;
      float c = (mp[16 + row] != 0) ? s1[r] : -1e30f;
      s0[r] = a; s1[r] = c;
      float t = fmaxf(a, c);
      t = fmaxf(t, __shfl_xor(t, 1, 16));
      t = fmaxf(t, __shfl_xor(t, 2, 16));
      t = fmaxf(t, __shfl_xor(t, 4, 16));
      t = fmaxf(t, __shfl_xor(t, 8, 16));
      mnew[r] = fmaxf(m_i[r], t);
    }

    // exp2, row sums, O rescale, spill P to LDS (C layout)
#pragma unroll
    for (int r = 0; r < 8; ++r) {
      const float alpha = exp2f(m_i[r] - mnew[r]);
      m_i[r] = mnew[r];
      const float p0 = exp2f(s0[r] - mnew[r]);
      const float p1 = exp2f(s1[r] - mnew[r]);
      float rs = p0 + p1;
      rs += __shfl_xor(rs, 1, 16);
      rs += __shfl_xor(rs, 2, 16);
      rs += __shfl_xor(rs, 4, 16);
      rs += __shfl_xor(rs, 8, 16);
      l_i[r] = l_i[r] * alpha + rs;
      P[(r + hi * 8) * 36 + row]      = p0;
      P[(r + hi * 8) * 36 + 16 + row] = p1;
      O[0][r] *= alpha; O[1][r] *= alpha; O[2][r] *= alpha; O[3][r] *= alpha;
    }

    // reload P in A layout (same-wave LDS ops are in-order; compiler waits dscnt)
    v16bf pf;
    {
      const float* pr = P + row * 36;
      v4f f0 = *(const v4f*)(pr + kbA);
      v4f f1 = *(const v4f*)(pr + kbA + 4);
      v4f f2 = *(const v4f*)(pr + 16 + kbA);
      v4f f3 = *(const v4f*)(pr + 16 + kbA + 4);
#pragma unroll
      for (int i = 0; i < 4; ++i) {
        pf[i]      = (__bf16)f0[i];
        pf[4 + i]  = (__bf16)f1[i];
        pf[8 + i]  = (__bf16)f2[i];
        pf[12 + i] = (__bf16)f3[i];
      }
    }

    // O += P(16x32) @ V(32x64); vhT gives contiguous-K B fragments
#pragma unroll
    for (int j = 0; j < 4; ++j) {
      const __bf16* vr = vhT + (size_t)(j * 16 + row) * kRows + b * kSeq + kt + kbB;
      O[j] = wmma_bf16(pf, b_frag_bf16(vr), O[j]);
    }
  }

  // finalize: O / l, store bf16 row-major [16384, 64]
#pragma unroll
  for (int j = 0; j < 4; ++j) {
#pragma unroll
    for (int r = 0; r < 8; ++r) {
      obf[(size_t)(b * kSeq + q0 + r + hi * 8) * kDK + j * 16 + row] =
          (__bf16)(O[j][r] / l_i[r]);
    }
  }
}

// ---------------------------------------------------------------------------
// Stage 3: Y = O @ Wo^T + bo -> f32 [16384, 1024]. 1024 wave-jobs.
// ---------------------------------------------------------------------------
__global__ __launch_bounds__(256) void out_proj_wmma(
    const __bf16* __restrict__ obf, const float* __restrict__ Wo,
    const float* __restrict__ bo, float* __restrict__ out) {
  const int lane = threadIdx.x & 31;
  const int wv   = threadIdx.x >> 5;
  const int job  = blockIdx.x * 8 + wv;     // 0..1023
  const int m0   = job * 16;
  const int row  = lane & 15;
  const int hi   = lane >> 4;
  const int kbA  = hi * 8;
  const int kbB  = hi * 8;

  const __bf16* orp = obf + (size_t)(m0 + row) * kDK;
  const v16bf a0 = a_frag_bf16(orp, kbA);        // K = 0..31
  const v16bf a1 = a_frag_bf16(orp + 32, kbA);   // K = 32..63

  for (int j = 0; j < 64; ++j) {
    const float* wr = Wo + (size_t)(j * 16 + row) * kDK + kbB; // B[k,n]=Wo[n,k]
    v8f acc = {};
    acc = wmma_bf16(a0, b_frag_f32(wr), acc);
    acc = wmma_bf16(a1, b_frag_f32(wr + 32), acc);
    const float bval = bo[j * 16 + row];
#pragma unroll
    for (int r = 0; r < 8; ++r) {
      out[(size_t)(m0 + r + hi * 8) * kDModel + j * 16 + row] = acc[r] + bval;
    }
  }
}

// ---------------------------------------------------------------------------
extern "C" void kernel_launch(void* const* d_in, const int* in_sizes, int n_in,
                              void* d_out, int out_size, void* d_ws, size_t ws_size,
                              hipStream_t stream) {
  const float* q  = (const float*)d_in[0];
  const float* k  = (const float*)d_in[1];
  const float* v  = (const float*)d_in[2];
  const int* mask = (const int*)d_in[3];
  const float* Wq = (const float*)d_in[4];
  const float* bq = (const float*)d_in[5];
  const float* Wk = (const float*)d_in[6];
  const float* bk = (const float*)d_in[7];
  const float* Wv = (const float*)d_in[8];
  const float* bv = (const float*)d_in[9];
  const float* Wo = (const float*)d_in[10];
  const float* bo = (const float*)d_in[11];

  // Workspace layout: 4 bf16 segments of 16384*64 elems (2MB each, 8MB total)
  char* ws = (char*)d_ws;
  const size_t SEG = (size_t)kRows * kDK * sizeof(__bf16);
  __bf16* qh  = (__bf16*)(ws);
  __bf16* kh  = (__bf16*)(ws + SEG);
  __bf16* vhT = (__bf16*)(ws + 2 * SEG);
  __bf16* obf = (__bf16*)(ws + 3 * SEG);

  proj_wmma<<<384, 256, 0, stream>>>(q, k, v, Wq, bq, Wk, bk, Wv, bv, qh, kh, vhT);
  flash_attn_wmma<<<128, 256, 0, stream>>>(qh, kh, vhT, mask, obf);
  out_proj_wmma<<<128, 256, 0, stream>>>(obf, Wo, bo, (float*)d_out);
}
